// StageNet_83485574300222
// MI455X (gfx1250) — matrix operand order repaired and checked by reading, not compiled, not run on hardware
//
#include <hip/hip_runtime.h>

// ---------------------------------------------------------------------------
// StageNet forward for MI455X (gfx1250, wave32, WMMA bf16 -> f32 accumulate)
// B=64, T=256, ICD=2000, H=384, L=3, CH=128, W=10
// All GEMMs: v_wmma_f32_16x16x32_bf16, wave computes 16x64 strip (A reused x4).
// ---------------------------------------------------------------------------

#define BATCH 64
#define TSTEPS 256
#define ICD 2000
#define HDIM 384
#define LLEV 3
#define CHNK 128
#define WIN 10
#define NXO 1542          // 4*H + 2*L
#define NXO_PAD 1600      // 100 * 16 (divisible by 64 for 4-tile waves)
#define KAUG 385          // H + 1
#define KAUG_PAD 416      // 13 * 32
#define KIN_PAD 2048      // K pad of ICD for embed GEMM (B rows >=2000 zero)
#define NOUT_PAD 2048     // 128 * 16 N pad for out GEMM
#define KCONV 3840        // H * W
#define BT (BATCH * TSTEPS)

typedef __attribute__((ext_vector_type(16))) __bf16 bf16x16;
typedef __attribute__((ext_vector_type(8)))  __bf16 bf16x8;
typedef __attribute__((ext_vector_type(8)))  float  f32x8;
typedef __attribute__((ext_vector_type(4)))  float  f32x4;

__device__ __forceinline__ f32x8 wmma_bf16(bf16x16 a, bf16x16 b, f32x8 c) {
  // (neg_a, A, neg_b, B, c_mod, C, reuse_a, reuse_b)
  return __builtin_amdgcn_wmma_f32_16x16x32_bf16(false, a, false, b, (short)0, c,
                                                 false, false);
}

// A fragment (16x32, row m = lane&15): two contiguous 8-elem K runs per lane
// (lanes 0-15: K {0..7,16..23}; lanes 16-31: K {8..15,24..31}).
__device__ __forceinline__ bf16x16 ldA_bf16(const __bf16* __restrict__ row,
                                            int k0, int kh) {
  const bf16x8* p = (const bf16x8*)(row + k0 + kh * 8);   // 16B aligned
  bf16x8 lo = p[0], hi = p[2];                            // +16 elements
  bf16x16 a;
#pragma unroll
  for (int i = 0; i < 8; ++i) { a[i] = lo[i]; a[i + 8] = hi[i]; }
  return a;
}

// Unguarded f32 A fragment via float4 loads (K must be padded/multiple of 32).
__device__ __forceinline__ bf16x16 ldA_f32(const float* __restrict__ row,
                                           int k0, int kh) {
  const f32x4* p = (const f32x4*)(row + k0 + kh * 8);     // 16B aligned
  f32x4 q0 = p[0], q1 = p[1], q2 = p[4], q3 = p[5];       // +16,+20 floats
  bf16x16 a;
#pragma unroll
  for (int i = 0; i < 4; ++i) {
    a[i]      = (__bf16)q0[i];
    a[i + 4]  = (__bf16)q1[i];
    a[i + 8]  = (__bf16)q2[i];
    a[i + 12] = (__bf16)q3[i];
  }
  return a;
}

// Tail chunk for embed GEMM (k0 = 1984, K = 2000): first 8-run is always
// in-bounds (1984+8+7 = 1999), second run is always past-end -> zero
// (B rows >= 2000 are zero-padded, so value is irrelevant but must be finite).
__device__ __forceinline__ bf16x16 ldA_f32_tail(const float* __restrict__ row,
                                                int kh) {
  const f32x4* p = (const f32x4*)(row + 1984 + kh * 8);
  f32x4 q0 = p[0], q1 = p[1];
  bf16x16 a;
#pragma unroll
  for (int i = 0; i < 4; ++i) {
    a[i]      = (__bf16)q0[i];
    a[i + 4]  = (__bf16)q1[i];
    a[i + 8]  = (__bf16)0.f;
    a[i + 12] = (__bf16)0.f;
  }
  return a;
}

// B fragment (32x16) from N-major weight copy wt[N][Kpitch]:
// lane n = lane&15 -> column n0+n, two 8-elem K runs (same K order as A).
__device__ __forceinline__ bf16x16 ldB(const __bf16* __restrict__ wt, int Kpitch,
                                       int n0, int k0, int lane) {
  const bf16x8* p = (const bf16x8*)(wt + (size_t)(n0 + (lane & 15)) * Kpitch +
                                    k0 + (lane >> 4) * 8);
  bf16x8 lo = p[0], hi = p[2];
  bf16x16 b;
#pragma unroll
  for (int i = 0; i < 8; ++i) { b[i] = lo[i]; b[i + 8] = hi[i]; }
  return b;
}

__device__ __forceinline__ float sigmoidf_(float x) { return 1.f / (1.f + expf(-x)); }

// ---------------------------------------------------------------------------
// Weight prep
// ---------------------------------------------------------------------------
__global__ void k_transpose_bf16(__bf16* __restrict__ dst, const float* __restrict__ src,
                                 int Ksrc, int Nsrc, int Kpitch, long total) {
  long i = (long)blockIdx.x * blockDim.x + threadIdx.x;
  if (i >= total) return;
  int n = (int)(i / Kpitch), k = (int)(i % Kpitch);
  float v = (k < Ksrc && n < Nsrc) ? src[(size_t)k * Nsrc + n] : 0.f;
  dst[i] = (__bf16)v;
}

__global__ void k_convert_bf16(__bf16* __restrict__ dst, const float* __restrict__ src,
                               long total) {
  long i = (long)blockIdx.x * blockDim.x + threadIdx.x;
  if (i < total) dst[i] = (__bf16)src[i];
}

__global__ void k_init(__bf16* __restrict__ embA, __bf16* __restrict__ h_aug,
                       float* __restrict__ cst, float* __restrict__ ring_h,
                       float* __restrict__ ring_d) {
  size_t t0 = (size_t)blockIdx.x * blockDim.x + threadIdx.x;
  size_t st = (size_t)gridDim.x * blockDim.x;
  for (size_t i = t0; i < (size_t)BT * 32; i += st) {           // embA cols 384..415
    size_t row = i >> 5; int col = HDIM + (int)(i & 31);
    embA[row * KAUG_PAD + col] = (__bf16)((col == HDIM) ? 1.f : 0.f);
  }
  for (size_t i = t0; i < (size_t)BATCH * KAUG_PAD; i += st) {  // h_aug = [0..,1,0..]
    int k = (int)(i % KAUG_PAD);
    h_aug[i] = (__bf16)((k == HDIM) ? 1.f : 0.f);
  }
  for (size_t i = t0; i < (size_t)BATCH * HDIM; i += st) cst[i] = 0.f;
  for (size_t i = t0; i < (size_t)BATCH * HDIM * WIN; i += st) ring_h[i] = 0.f;
  for (size_t i = t0; i < (size_t)BATCH * WIN; i += st) ring_d[i] = 0.f;
}

// ---------------------------------------------------------------------------
// GEMM 1: embed = x @ dia_w + dia_b -> augmented bf16 activation embA.
// Wave computes 16x64 strip (4 N-tiles, A reused). 62 unguarded K chunks + tail.
// ---------------------------------------------------------------------------
__global__ void k_gemm_embed(const float* __restrict__ x, const __bf16* __restrict__ wt,
                             const float* __restrict__ bias, __bf16* __restrict__ embA) {
  int gw = (int)((blockIdx.x * blockDim.x + threadIdx.x) >> 5);
  int lane = threadIdx.x & 31;
  const int ntG = HDIM / 64;                      // 6 groups of 4 tiles
  if (gw >= (BT / 16) * ntG) return;
  int tm = gw / ntG, tg = gw % ntG;
  int row0 = tm * 16, n0 = tg * 64;
  int m = lane & 15, kh = lane >> 4;
  const float* arow = x + (size_t)(row0 + m) * ICD;
  f32x8 c0 = {}, c1 = {}, c2 = {}, c3 = {};
  for (int k0 = 0; k0 < 1984; k0 += 32) {
    __builtin_prefetch(arow + k0 + 256, 0, 3);    // global_prefetch_b8
    bf16x16 a = ldA_f32(arow, k0, kh);
    c0 = wmma_bf16(a, ldB(wt, KIN_PAD, n0,      k0, lane), c0);
    c1 = wmma_bf16(a, ldB(wt, KIN_PAD, n0 + 16, k0, lane), c1);
    c2 = wmma_bf16(a, ldB(wt, KIN_PAD, n0 + 32, k0, lane), c2);
    c3 = wmma_bf16(a, ldB(wt, KIN_PAD, n0 + 48, k0, lane), c3);
  }
  { // tail chunk k0 = 1984 (valid K up to 1999; B rows >=2000 are zero)
    bf16x16 a = ldA_f32_tail(arow, kh);
    c0 = wmma_bf16(a, ldB(wt, KIN_PAD, n0,      1984, lane), c0);
    c1 = wmma_bf16(a, ldB(wt, KIN_PAD, n0 + 16, 1984, lane), c1);
    c2 = wmma_bf16(a, ldB(wt, KIN_PAD, n0 + 32, 1984, lane), c2);
    c3 = wmma_bf16(a, ldB(wt, KIN_PAD, n0 + 48, 1984, lane), c3);
  }
  f32x8 cc[4] = {c0, c1, c2, c3};
#pragma unroll
  for (int j = 0; j < 4; ++j) {
    int n = n0 + j * 16 + (lane & 15);
    float bv = bias[n];
#pragma unroll
    for (int r = 0; r < 8; ++r) {
      int mm = row0 + (lane >> 4) * 8 + r;
      embA[(size_t)mm * KAUG_PAD + n] = (__bf16)(cc[j][r] + bv);
    }
  }
}

// ---------------------------------------------------------------------------
// GEMM 2: kx = embA @ kernel_w + (kernel_b + rec_b)   (f32, BT x 1542)
// ---------------------------------------------------------------------------
__global__ void k_gemm_kx(const __bf16* __restrict__ embA, const __bf16* __restrict__ wt,
                          const float* __restrict__ kb, const float* __restrict__ rb,
                          float* __restrict__ kx) {
  int gw = (int)((blockIdx.x * blockDim.x + threadIdx.x) >> 5);
  int lane = threadIdx.x & 31;
  const int ntG = NXO_PAD / 64;                   // 25 groups
  if (gw >= (BT / 16) * ntG) return;
  int tm = gw / ntG, tg = gw % ntG;
  int row0 = tm * 16, n0 = tg * 64;
  int m = lane & 15, kh = lane >> 4;
  const __bf16* arow = embA + (size_t)(row0 + m) * KAUG_PAD;
  f32x8 c0 = {}, c1 = {}, c2 = {}, c3 = {};
  for (int k0 = 0; k0 < KAUG_PAD; k0 += 32) {
    bf16x16 a = ldA_bf16(arow, k0, kh);
    c0 = wmma_bf16(a, ldB(wt, KAUG_PAD, n0,      k0, lane), c0);
    c1 = wmma_bf16(a, ldB(wt, KAUG_PAD, n0 + 16, k0, lane), c1);
    c2 = wmma_bf16(a, ldB(wt, KAUG_PAD, n0 + 32, k0, lane), c2);
    c3 = wmma_bf16(a, ldB(wt, KAUG_PAD, n0 + 48, k0, lane), c3);
  }
  f32x8 cc[4] = {c0, c1, c2, c3};
#pragma unroll
  for (int j = 0; j < 4; ++j) {
    int n = n0 + j * 16 + (lane & 15);
    if (n < NXO) {
      float bv = kb[n] + rb[n];
#pragma unroll
      for (int r = 0; r < 8; ++r) {
        int mm = row0 + (lane >> 4) * 8 + r;
        kx[(size_t)mm * NXO + n] = cc[j][r] + bv;
      }
    }
  }
}

// ---------------------------------------------------------------------------
// Per-step S1: xo = kx[:,t,:] + [h,1] @ rec_w   (64 x 1600, f32)
// ---------------------------------------------------------------------------
__global__ void k_gemm_rec(const __bf16* __restrict__ h_aug, const __bf16* __restrict__ wt,
                           const float* __restrict__ kx, float* __restrict__ xo, int t) {
  int gw = (int)((blockIdx.x * blockDim.x + threadIdx.x) >> 5);
  int lane = threadIdx.x & 31;
  const int ntG = NXO_PAD / 64;                   // 25 groups
  if (gw >= (BATCH / 16) * ntG) return;           // 100 waves
  int tm = gw / ntG, tg = gw % ntG;
  int row0 = tm * 16, n0 = tg * 64;
  int m = lane & 15, kh = lane >> 4;
  const __bf16* arow = h_aug + (size_t)(row0 + m) * KAUG_PAD;
  f32x8 c0 = {}, c1 = {}, c2 = {}, c3 = {};
  for (int k0 = 0; k0 < KAUG_PAD; k0 += 32) {
    bf16x16 a = ldA_bf16(arow, k0, kh);
    c0 = wmma_bf16(a, ldB(wt, KAUG_PAD, n0,      k0, lane), c0);
    c1 = wmma_bf16(a, ldB(wt, KAUG_PAD, n0 + 16, k0, lane), c1);
    c2 = wmma_bf16(a, ldB(wt, KAUG_PAD, n0 + 32, k0, lane), c2);
    c3 = wmma_bf16(a, ldB(wt, KAUG_PAD, n0 + 48, k0, lane), c3);
  }
  f32x8 cc[4] = {c0, c1, c2, c3};
#pragma unroll
  for (int j = 0; j < 4; ++j) {
    int n = n0 + j * 16 + (lane & 15);
#pragma unroll
    for (int r = 0; r < 8; ++r) {
      int b = row0 + (lane >> 4) * 8 + r;
      float add = (n < NXO) ? kx[((size_t)b * TSTEPS + t) * NXO + n] : 0.f;
      xo[(size_t)b * NXO_PAD + n] = cc[j][r] + add;
    }
  }
}

// ---------------------------------------------------------------------------
// Per-step S2+S3 fused: gates -> state update -> ring push -> local_dis ->
// theme MLP -> conv (WMMA from LDS-staged A, A reused across 3 tiles) ->
// rnn[:,t,:] = h_new + theme * conv.   16 blocks x 256 thr (4 batches/block).
// ---------------------------------------------------------------------------
#define SB 4
__global__ void k_step(const float* __restrict__ xo, float* __restrict__ cst,
                       float* __restrict__ ring_h, float* __restrict__ ring_d,
                       __bf16* __restrict__ h_aug, const __bf16* __restrict__ conv_wb,
                       const float* __restrict__ scale_w, const float* __restrict__ scale_b,
                       const float* __restrict__ rescale_w, const float* __restrict__ rescale_b,
                       const float* __restrict__ conv_b, float* __restrict__ rnn, int t) {
  __shared__ float  fmim[SB][2][LLEV];
  __shared__ float  dbuf[SB];
  __shared__ float  hbuf[SB][HDIM];
  __shared__ float  ldis[SB][WIN];
  __shared__ float  mh[SB][HDIM];
  __shared__ float  th1[SB][HDIM / 6];
  __shared__ float  theme[SB][HDIM];
  __shared__ __bf16 Abuf[SB][KCONV];

  int b0 = blockIdx.x * SB;
  int tid = threadIdx.x;
  int slotNew = t % WIN;

  // --- stage 0: master gates fm/im + cur_dis (one thread per batch) ---
  if (tid < SB) {
    int b = b0 + tid;
    const float* xr = xo + (size_t)b * NXO_PAD;
    float a0 = xr[0], a1 = xr[1], a2 = xr[2];
    float mx = fmaxf(a0, fmaxf(a1, a2));
    float e0 = expf(a0 - mx), e1 = expf(a1 - mx), e2 = expf(a2 - mx);
    float s = e0 + e1 + e2;
    float f0 = e0 / s, f1 = (e0 + e1) / s, f2 = 1.f;   // cumax l2r
    fmim[tid][0][0] = f0; fmim[tid][0][1] = f1; fmim[tid][0][2] = f2;
    float c0 = xr[3], c1 = xr[4], c2 = xr[5];
    mx = fmaxf(c0, fmaxf(c1, c2));
    e0 = expf(c0 - mx); e1 = expf(c1 - mx); e2 = expf(c2 - mx);
    s = e0 + e1 + e2;
    float p1 = e1 / s, p2 = e2 / s;
    fmim[tid][1][0] = 1.f; fmim[tid][1][1] = p1 + p2; fmim[tid][1][2] = p2;
    float cd = 1.f - (f0 + f1 + f2) * (1.f / 3.f);
    dbuf[tid] = cd;
    ring_d[b * WIN + slotNew] = cd;
  }
  __syncthreads();

  // --- stage 1: elementwise LSTM gates + state update ---
  for (int e = tid; e < SB * HDIM; e += 256) {
    int bl = e / HDIM, hh = e % HDIM;
    int b = b0 + bl;
    int l = hh >> 7, ch = hh & 127;
    const float* xr = xo + (size_t)b * NXO_PAD;
    float fmv = fmim[bl][0][l], imv = fmim[bl][1][l];
    float fv = sigmoidf_(xr[6 + l * CHNK + ch]);
    float iv = sigmoidf_(xr[6 + HDIM + l * CHNK + ch]);
    float og = sigmoidf_(xr[6 + 2 * HDIM + l * CHNK + ch]);
    float ci = tanhf(xr[6 + 3 * HDIM + l * CHNK + ch]);
    float cl = cst[(size_t)b * HDIM + hh];
    float ov = fmv * imv;
    float cn = ov * (fv * cl + iv * ci) + (fmv - ov) * cl + (imv - ov) * ci;
    float hn = og * tanhf(cn);
    cst[(size_t)b * HDIM + hh] = cn;
    hbuf[bl][hh] = hn;
    ring_h[((size_t)b * HDIM + hh) * WIN + slotNew] = hn;
    h_aug[(size_t)b * KAUG_PAD + hh] = (__bf16)hn;
  }
  __syncthreads();

  // --- stage 2: local_dis = softmax(cumsum(ring_dis)) per batch ---
  if (tid < SB) {
    int b = b0 + tid;
    float cum = 0.f, vals[WIN];
#pragma unroll
    for (int w = 0; w < WIN; ++w) {
      int slot = (t + 1 + w) % WIN;
      float dv = (w == WIN - 1) ? dbuf[tid] : ring_d[b * WIN + slot];
      cum += dv; vals[w] = cum;
    }
    float mx = vals[0];
#pragma unroll
    for (int w = 1; w < WIN; ++w) mx = fmaxf(mx, vals[w]);
    float s = 0.f;
#pragma unroll
    for (int w = 0; w < WIN; ++w) { vals[w] = expf(vals[w] - mx); s += vals[w]; }
    float inv = 1.f / s;
#pragma unroll
    for (int w = 0; w < WIN; ++w) ldis[tid][w] = vals[w] * inv;
  }
  __syncthreads();

  // --- stage 3: build conv A matrix (bf16, K = h*W+w) + mean over W ---
  for (int e = tid; e < SB * HDIM; e += 256) {
    int bl = e / HDIM, hh = e % HDIM;
    int b = b0 + bl;
    const float* rh = ring_h + ((size_t)b * HDIM + hh) * WIN;
    float acc = 0.f;
#pragma unroll
    for (int w = 0; w < WIN; ++w) {
      int slot = (t + 1 + w) % WIN;
      float hv = (w == WIN - 1) ? hbuf[bl][hh] : rh[slot];
      float lv = hv * ldis[bl][w];
      Abuf[bl][hh * WIN + w] = (__bf16)lv;
      acc += lv;
    }
    mh[bl][hh] = acc * (1.f / WIN);
  }
  __syncthreads();

  // --- stage 4: theme MLP (384 -> 64 relu -> 384 sigmoid) ---
  for (int e = tid; e < SB * (HDIM / 6); e += 256) {
    int bl = e / (HDIM / 6), n = e % (HDIM / 6);
    float acc = scale_b[n];
    for (int h = 0; h < HDIM; ++h) acc += mh[bl][h] * scale_w[h * (HDIM / 6) + n];
    th1[bl][n] = fmaxf(acc, 0.f);
  }
  __syncthreads();
  for (int e = tid; e < SB * HDIM; e += 256) {
    int bl = e / HDIM, n = e % HDIM;
    float acc = rescale_b[n];
    for (int h = 0; h < HDIM / 6; ++h) acc += th1[bl][h] * rescale_w[h * HDIM + n];
    theme[bl][n] = sigmoidf_(acc);
  }
  __syncthreads();

  // --- stage 5: conv einsum as WMMA GEMM (A from LDS, reused over 3 tiles) ---
  int wv = tid >> 5, lane = tid & 31;
  int m = lane & 15, kh = lane >> 4;
  bf16x16 zeroA;
#pragma unroll
  for (int i = 0; i < 16; ++i) zeroA[i] = (__bf16)0.f;
  f32x8 cc0 = {}, cc1 = {}, cc2 = {};
  int nb = wv * 48;                                // 3 tiles per wave
  for (int k0 = 0; k0 < KCONV; k0 += 32) {
    bf16x16 a = (m < SB) ? ldA_bf16(&Abuf[m][0], k0, kh) : zeroA;
    cc0 = wmma_bf16(a, ldB(conv_wb, KCONV, nb,      k0, lane), cc0);
    cc1 = wmma_bf16(a, ldB(conv_wb, KCONV, nb + 16, k0, lane), cc1);
    cc2 = wmma_bf16(a, ldB(conv_wb, KCONV, nb + 32, k0, lane), cc2);
  }
  f32x8 cc[3] = {cc0, cc1, cc2};
#pragma unroll
  for (int tt = 0; tt < 3; ++tt) {
#pragma unroll
    for (int r = 0; r < 8; ++r) {
      int mrow = (lane >> 4) * 8 + r;
      if (mrow < SB) {
        int o = nb + tt * 16 + (lane & 15);
        int b = b0 + mrow;
        float convv = cc[tt][r] + conv_b[o];
        rnn[((size_t)b * TSTEPS + t) * HDIM + o] = hbuf[mrow][o] + theme[mrow][o] * convv;
      }
    }
  }
}

// ---------------------------------------------------------------------------
// GEMM 3: out = rnn @ out_w + out_b  (16384 x 384 x 2000, N padded to 2048)
// ---------------------------------------------------------------------------
__global__ void k_gemm_out(const float* __restrict__ rnn, const __bf16* __restrict__ wt,
                           const float* __restrict__ bias, float* __restrict__ out) {
  int gw = (int)((blockIdx.x * blockDim.x + threadIdx.x) >> 5);
  int lane = threadIdx.x & 31;
  const int ntG = NOUT_PAD / 64;                  // 32 groups
  if (gw >= (BT / 16) * ntG) return;
  int tm = gw / ntG, tg = gw % ntG;
  int row0 = tm * 16, n0 = tg * 64;
  int m = lane & 15, kh = lane >> 4;
  const float* arow = rnn + (size_t)(row0 + m) * HDIM;
  f32x8 c0 = {}, c1 = {}, c2 = {}, c3 = {};
  for (int k0 = 0; k0 < HDIM; k0 += 32) {         // K = 384 = 12*32, no guards
    bf16x16 a = ldA_f32(arow, k0, kh);
    c0 = wmma_bf16(a, ldB(wt, HDIM, n0,      k0, lane), c0);
    c1 = wmma_bf16(a, ldB(wt, HDIM, n0 + 16, k0, lane), c1);
    c2 = wmma_bf16(a, ldB(wt, HDIM, n0 + 32, k0, lane), c2);
    c3 = wmma_bf16(a, ldB(wt, HDIM, n0 + 48, k0, lane), c3);
  }
  f32x8 cc[4] = {c0, c1, c2, c3};
#pragma unroll
  for (int j = 0; j < 4; ++j) {
    int n = n0 + j * 16 + (lane & 15);
    if (n < ICD) {
      float bv = bias[n];
#pragma unroll
      for (int r = 0; r < 8; ++r) {
        int mm = row0 + (lane >> 4) * 8 + r;
        out[(size_t)mm * ICD + n] = cc[j][r] + bv;
      }
    }
  }
}

// ---------------------------------------------------------------------------
// Host launcher
// ---------------------------------------------------------------------------
extern "C" void kernel_launch(void* const* d_in, const int* in_sizes, int n_in,
                              void* d_out, int out_size, void* d_ws, size_t ws_size,
                              hipStream_t stream) {
  const float* x        = (const float*)d_in[0];
  const float* dia_w    = (const float*)d_in[1];
  const float* dia_b    = (const float*)d_in[2];
  const float* kernel_w = (const float*)d_in[3];
  const float* kernel_b = (const float*)d_in[4];
  const float* rec_w    = (const float*)d_in[5];
  const float* rec_b    = (const float*)d_in[6];
  const float* scale_w  = (const float*)d_in[7];
  const float* scale_b  = (const float*)d_in[8];
  const float* rescale_w = (const float*)d_in[9];
  const float* rescale_b = (const float*)d_in[10];
  const float* conv_w   = (const float*)d_in[11];
  const float* conv_b   = (const float*)d_in[12];
  const float* out_w    = (const float*)d_in[13];
  const float* out_b    = (const float*)d_in[14];
  float* out = (float*)d_out;

  // workspace bump allocator (~152 MB)
  char* wsp = (char*)d_ws;
  auto alloc = [&](size_t bytes) {
    char* p = wsp; wsp += (bytes + 255) & ~(size_t)255; return p;
  };
  __bf16* dia_wT    = (__bf16*)alloc((size_t)HDIM * KIN_PAD * 2);
  __bf16* kernel_wT = (__bf16*)alloc((size_t)NXO_PAD * KAUG_PAD * 2);
  __bf16* rec_wT    = (__bf16*)alloc((size_t)NXO_PAD * KAUG_PAD * 2);
  __bf16* out_wT    = (__bf16*)alloc((size_t)NOUT_PAD * HDIM * 2);
  __bf16* conv_wb   = (__bf16*)alloc((size_t)HDIM * KCONV * 2);
  __bf16* embA      = (__bf16*)alloc((size_t)BT * KAUG_PAD * 2);
  float*  kx        = (float*)alloc((size_t)BT * NXO * 4);
  float*  xo        = (float*)alloc((size_t)BATCH * NXO_PAD * 4);
  __bf16* h_aug     = (__bf16*)alloc((size_t)BATCH * KAUG_PAD * 2);
  float*  cst       = (float*)alloc((size_t)BATCH * HDIM * 4);
  float*  ring_h    = (float*)alloc((size_t)BATCH * HDIM * WIN * 4);
  float*  ring_d    = (float*)alloc((size_t)BATCH * WIN * 4);
  float*  rnn       = (float*)alloc((size_t)BT * HDIM * 4);

  auto blocks1d = [](long total) { return (unsigned)((total + 255) / 256); };

  // weight prep
  { long tot = (long)HDIM * KIN_PAD;
    k_transpose_bf16<<<blocks1d(tot), 256, 0, stream>>>(dia_wT, dia_w, ICD, HDIM, KIN_PAD, tot); }
  { long tot = (long)NXO_PAD * KAUG_PAD;
    k_transpose_bf16<<<blocks1d(tot), 256, 0, stream>>>(kernel_wT, kernel_w, KAUG, NXO, KAUG_PAD, tot);
    k_transpose_bf16<<<blocks1d(tot), 256, 0, stream>>>(rec_wT, rec_w, KAUG, NXO, KAUG_PAD, tot); }
  { long tot = (long)NOUT_PAD * HDIM;
    k_transpose_bf16<<<blocks1d(tot), 256, 0, stream>>>(out_wT, out_w, HDIM, ICD, HDIM, tot); }
  { long tot = (long)HDIM * KCONV;
    k_convert_bf16<<<blocks1d(tot), 256, 0, stream>>>(conv_wb, conv_w, tot); }

  // state / augmentation init (re-run every call: deterministic under replay)
  k_init<<<1024, 256, 0, stream>>>(embA, h_aug, cst, ring_h, ring_d);

  // embed GEMM: 1024 * 6 wave-strips, 8 waves/block
  k_gemm_embed<<<(BT / 16) * (HDIM / 64) / 8, 256, 0, stream>>>(x, dia_wT, dia_b, embA);

  // kx GEMM: 1024 * 25 wave-strips
  k_gemm_kx<<<(BT / 16) * (NXO_PAD / 64) / 8, 256, 0, stream>>>(embA, kernel_wT,
                                                                kernel_b, rec_b, kx);

  // sequential scan: 2 kernels per step (kernel boundary = global sync)
  const int recWaves = (BATCH / 16) * (NXO_PAD / 64);   // 100
  const unsigned recBlocks = (recWaves + 7) / 8;        // 13
  for (int t = 0; t < TSTEPS; ++t) {
    k_gemm_rec<<<recBlocks, 256, 0, stream>>>(h_aug, rec_wT, kx, xo, t);
    k_step<<<BATCH / SB, 256, 0, stream>>>(xo, cst, ring_h, ring_d, h_aug, conv_wb,
                                           scale_w, scale_b, rescale_w, rescale_b,
                                           conv_b, rnn, t);
  }

  // output GEMM: 1024 * 32 wave-strips
  k_gemm_out<<<(BT / 16) * (NOUT_PAD / 64) / 8, 256, 0, stream>>>(rnn, out_wT, out_b, out);
}
